// MultiHeadAttention_25915832664656
// MI455X (gfx1250) — matrix operand order repaired and checked by reading, not compiled, not run on hardware
//
#include <hip/hip_runtime.h>

// Problem constants (from reference): B=2, S=2048, D=1024, H=16, DH=64
#define BB   2
#define SS   2048
#define DD   1024
#define HH   16
#define DHH  64
#define MTOT (BB * SS)   // 4096 rows of the [B*S, D] activation matrices

typedef __bf16 bf16;
typedef __attribute__((ext_vector_type(16))) __bf16 v16bf;
typedef __attribute__((ext_vector_type(8)))  __bf16 v8bf;
typedef __attribute__((ext_vector_type(8)))  float  v8f;
typedef __attribute__((ext_vector_type(4)))  int    v4i;

#ifndef __has_builtin
#define __has_builtin(x) 0
#endif
#if __has_builtin(__builtin_amdgcn_global_load_async_to_lds_b128) && \
    __has_builtin(__builtin_amdgcn_s_wait_asynccnt)
#define HAVE_ASYNC 1
#else
#define HAVE_ASYNC 0
#endif

#define AS1 __attribute__((address_space(1)))
#define AS3 __attribute__((address_space(3)))

// ---------------------------------------------------------------------------
// helpers
// ---------------------------------------------------------------------------

// float -> bf16 round-to-nearest-even (bit manipulation).
__device__ __forceinline__ bf16 f2bf(float f) {
  unsigned u = __float_as_uint(f);
  u += 0x7FFFu + ((u >> 16) & 1u);
  unsigned short hs = (unsigned short)(u >> 16);
  return __builtin_bit_cast(__bf16, hs);
}

__device__ __forceinline__ v8f wmma_bf16(v16bf a, v16bf b, v8f c) {
  return __builtin_amdgcn_wmma_f32_16x16x32_bf16(
      false, a, false, b, (short)0, c, false, false);
}

// 16-byte global -> LDS copy: async DMA (ASYNCcnt) when available,
// else a VGPR bounce.
__device__ __forceinline__ void async_copy16(const bf16* gsrc, bf16* ldst) {
#if HAVE_ASYNC
  // builtin signature (from probe diagnostic): (v4i AS1* src, v4i AS3* dst,
  // imm offset, imm cpol)
  __builtin_amdgcn_global_load_async_to_lds_b128(
      (AS1 v4i*)(void*)const_cast<bf16*>(gsrc),
      (AS3 v4i*)(void*)ldst, 0, 0);
#else
  *(v8bf*)ldst = *(const v8bf*)gsrc;
#endif
}
__device__ __forceinline__ void async_wait_all() {
#if HAVE_ASYNC
  __builtin_amdgcn_s_wait_asynccnt(0);
#endif
}

// A fragment: 16x32 bf16 tile from row-major memory (leading dim ld, elems).
// lane<16: row=lane, K={0..7,16..23}; lane>=16: row=lane-16, K={8..15,24..31}.
__device__ __forceinline__ v16bf load_frag_a(const bf16* base, int ld) {
  const int lane = threadIdx.x & 31;
  const int row  = lane & 15;
  const int kb   = (lane >> 4) << 3;   // 0 or 8
  const v8bf* p0 = (const v8bf*)(base + row * ld + kb);
  const v8bf* p1 = (const v8bf*)(base + row * ld + 16 + kb);
  v8bf a = *p0, b = *p1;
  v16bf f;
#pragma unroll
  for (int i = 0; i < 8; ++i) { f[i] = a[i]; f[8 + i] = b[i]; }
  return f;
}

// B fragment: 32x16 tile consumed from B^T stored row-major [N][K] (ld elems).
// lane<16: col=lane, K={0..7,16..23}; lane>=16: col=lane-16, K={8..15,24..31}.
__device__ __forceinline__ v16bf load_frag_b(const bf16* bt, int ld) {
  const int lane = threadIdx.x & 31;
  const int col  = lane & 15;
  const int kb   = (lane >> 4) << 3;   // 0 or 8
  const v8bf* p  = (const v8bf*)(bt + col * ld + kb);
  v8bf a = p[0], b = p[2];
  v16bf f;
#pragma unroll
  for (int i = 0; i < 8; ++i) { f[i] = a[i]; f[8 + i] = b[i]; }
  return f;
}

__device__ __forceinline__ float half16_max(float v) {
  v = fmaxf(v, __shfl_xor(v, 1, 32));
  v = fmaxf(v, __shfl_xor(v, 2, 32));
  v = fmaxf(v, __shfl_xor(v, 4, 32));
  v = fmaxf(v, __shfl_xor(v, 8, 32));
  return v;
}
__device__ __forceinline__ float half16_sum(float v) {
  v += __shfl_xor(v, 1, 32);
  v += __shfl_xor(v, 2, 32);
  v += __shfl_xor(v, 4, 32);
  v += __shfl_xor(v, 8, 32);
  return v;
}

// ---------------------------------------------------------------------------
// prep kernels
// ---------------------------------------------------------------------------

__global__ void cvt_bf16_kernel(const float* __restrict__ x,
                                bf16* __restrict__ y, int n) {
  int i = blockIdx.x * 256 + threadIdx.x;
  if (i < n) y[i] = f2bf(x[i]);
}

// wt[n][k] = (bf16) w[k][n]
__global__ void transpose_bf16_kernel(const float* __restrict__ w,
                                      bf16* __restrict__ wt) {
  int i = blockIdx.x * 256 + threadIdx.x;
  int n = i >> 10;
  int k = i & 1023;
  wt[i] = f2bf(w[k * DD + n]);
}

// ---------------------------------------------------------------------------
// projection GEMM with async-LDS double-buffered staging.
// Block = 256 threads (8 waves) computes a 128x128 tile; wave (wm,wn) does
// 32x64. A/B 128x32 bf16 panels staged per K-step.
// vmode 0: write [B,H,S,DH] (Q,K). vmode 1: write [B,H,DH,S] (V^T).
// ---------------------------------------------------------------------------
__global__ __launch_bounds__(256)
void proj_kernel(const bf16* __restrict__ X, const bf16* __restrict__ Wt,
                 const float* __restrict__ bias, bf16* __restrict__ out,
                 int vmode) {
  const int tiles_n = DD / 128;           // 8
  const int tm   = blockIdx.x / tiles_n;  // 0..31
  const int tn   = blockIdx.x % tiles_n;  // 0..7
  const int tid  = threadIdx.x;
  const int wid  = tid >> 5;
  const int wm   = wid >> 1;              // 0..3
  const int wn   = wid & 1;               // 0..1
  const int lane = tid & 31;
  const int col  = lane & 15;
  const int rbase = (lane >> 4) << 3;

  __shared__ __align__(16) bf16 Abuf[2][128 * 40];
  __shared__ __align__(16) bf16 Bbuf[2][128 * 40];

  const bf16* Ag = X  + (size_t)(tm * 128) * DD;
  const bf16* Bg = Wt + (size_t)(tn * 128) * DD;

  auto stage = [&](int k0, int bufi) {
#pragma unroll
    for (int i = 0; i < 2; ++i) {
      const int c   = tid + 256 * i;      // 0..511 chunk id
      const int row = c >> 2;             // 128 rows, 4x16B chunks per row
      const int off = (c & 3) * 8;        // bf16 element offset
      async_copy16(Ag + (size_t)row * DD + k0 + off,
                   &Abuf[bufi][row * 40 + off]);
      async_copy16(Bg + (size_t)row * DD + k0 + off,
                   &Bbuf[bufi][row * 40 + off]);
    }
  };

  v8f acc[2][4];
#pragma unroll
  for (int i = 0; i < 2; ++i)
#pragma unroll
    for (int j = 0; j < 4; ++j) {
      v8f z = {};
      acc[i][j] = z;
    }

  stage(0, 0);
  int buf = 0;
  for (int k0 = 0; k0 < DD; k0 += 32) {
    async_wait_all();     // my async loads for current buf done
    __syncthreads();      // everyone's are done; prev-buf readers finished
    if (k0 + 32 < DD) stage(k0 + 32, buf ^ 1);

    const bf16* Abase = &Abuf[buf][(wm * 32) * 40];
    const bf16* Bbase = &Bbuf[buf][(wn * 64) * 40];
    v16bf a0 = load_frag_a(Abase, 40);
    v16bf a1 = load_frag_a(Abase + 16 * 40, 40);
#pragma unroll
    for (int j = 0; j < 4; ++j) {
      v16bf bfrag = load_frag_b(Bbase + (j * 16) * 40, 40);
      acc[0][j] = wmma_bf16(a0, bfrag, acc[0][j]);
      acc[1][j] = wmma_bf16(a1, bfrag, acc[1][j]);
    }
    buf ^= 1;
  }

#pragma unroll
  for (int j = 0; j < 4; ++j) {
    const int n  = tn * 128 + wn * 64 + j * 16 + col;
    const float bv = bias[n];
    const int h  = n >> 6;
    const int dh = n & 63;
#pragma unroll
    for (int i = 0; i < 2; ++i) {
#pragma unroll
      for (int r = 0; r < 8; ++r) {
        const int m = tm * 128 + wm * 32 + i * 16 + rbase + r;
        const int b = m >> 11;
        const int s = m & 2047;
        const float v = acc[i][j][r] + bv;
        if (vmode == 0)
          out[((size_t)((b * HH + h) * SS + s)) * DHH + dh] = f2bf(v);
        else
          out[((size_t)((b * HH + h) * DHH + dh)) * SS + s] = f2bf(v);
      }
    }
  }
}

// ---------------------------------------------------------------------------
// flash attention. Block = 128 threads (4 waves) all on the SAME (b,h);
// wave w owns q-tile (blockIdx%16)*4 + w (32 queries). K/V 32-key panels are
// staged to LDS with async DMA (double buffered) and shared by all 4 waves.
// ---------------------------------------------------------------------------
__global__ __launch_bounds__(128)
void attn_kernel(const bf16* __restrict__ Q, const bf16* __restrict__ K,
                 const bf16* __restrict__ Vt, const int* __restrict__ mask,
                 float* __restrict__ Hout) {
  const int tid   = threadIdx.x;
  const int wid   = tid >> 5;
  const int bh    = blockIdx.x >> 4;          // 0..31
  const int qt    = (blockIdx.x & 15) * 4 + wid;  // 0..63
  const int b     = bh >> 4;
  const int h     = bh & 15;
  const int lane  = tid & 31;
  const int col   = lane & 15;
  const int rbase = (lane >> 4) << 3;

  __shared__ __align__(16) bf16 Kbuf[2][32 * 72];  // [key][dh] panel (+pad)
  __shared__ __align__(16) bf16 Vbuf[2][64 * 40];  // [dh][key] panel (+pad)
  __shared__ __align__(16) bf16 Pbuf[4][32][40];   // per-wave P tile
  __shared__ float mrow[4][32];
  __shared__ float lrow[4][32];

  const bf16* Qp = Q  + ((size_t)bh * SS + (size_t)qt * 32) * DHH;
  const bf16* Kp = K  + (size_t)bh * SS * DHH;
  const bf16* Vp = Vt + (size_t)bh * DHH * SS;
  const int*  mp = mask + b * SS;

  auto stage_kv = [&](int kb0, int bufi) {
#pragma unroll
    for (int i = 0; i < 2; ++i) {
      const int c = tid + 128 * i;            // 0..255
      const int krow = c >> 3;                // 32 rows, 8x16B chunks (64 bf16)
      const int koff = (c & 7) * 8;
      async_copy16(Kp + (size_t)(kb0 + krow) * DHH + koff,
                   &Kbuf[bufi][krow * 72 + koff]);
      const int vrow = c >> 2;                // 64 rows, 4x16B chunks (32 bf16)
      const int voff = (c & 3) * 8;
      async_copy16(Vp + (size_t)vrow * SS + kb0 + voff,
                   &Vbuf[bufi][vrow * 40 + voff]);
    }
  };

  mrow[wid][lane] = -1e30f;
  lrow[wid][lane] = 0.0f;

  // Q fragments are loop-invariant: hoist (4 x v16bf registers).
  v16bf qa[2][2];
#pragma unroll
  for (int k = 0; k < 2; ++k) {
    qa[k][0] = load_frag_a(Qp + k * 32, DHH);
    qa[k][1] = load_frag_a(Qp + 16 * DHH + k * 32, DHH);
  }

  v8f o[2][4];
#pragma unroll
  for (int i = 0; i < 2; ++i)
#pragma unroll
    for (int j = 0; j < 4; ++j) {
      v8f z = {};
      o[i][j] = z;
    }

  const float scale = 0.125f;   // 1/sqrt(DH)

  stage_kv(0, 0);
  int buf = 0;
  for (int kb0 = 0; kb0 < SS; kb0 += 32) {
    async_wait_all();
    __syncthreads();
    if (kb0 + 32 < SS) stage_kv(kb0 + 32, buf ^ 1);

    // ---- S = Q(32x64) @ K_block^T(64x32) from LDS K panel ----
    v8f sc[2][2];
#pragma unroll
    for (int i = 0; i < 2; ++i)
#pragma unroll
      for (int j = 0; j < 2; ++j) {
        v8f z = {};
        sc[i][j] = z;
      }
#pragma unroll
    for (int k = 0; k < 2; ++k) {
      v16bf b0 = load_frag_b(&Kbuf[buf][0 * 72] + k * 32, 72);
      v16bf b1 = load_frag_b(&Kbuf[buf][16 * 72] + k * 32, 72);
      sc[0][0] = wmma_bf16(qa[k][0], b0, sc[0][0]);
      sc[0][1] = wmma_bf16(qa[k][0], b1, sc[0][1]);
      sc[1][0] = wmma_bf16(qa[k][1], b0, sc[1][0]);
      sc[1][1] = wmma_bf16(qa[k][1], b1, sc[1][1]);
    }

    // ---- scale + additive mask ----
    float mk[2];
#pragma unroll
    for (int j = 0; j < 2; ++j)
      mk[j] = -10000.0f * (1.0f - (float)mp[kb0 + j * 16 + col]);
#pragma unroll
    for (int i = 0; i < 2; ++i)
#pragma unroll
      for (int j = 0; j < 2; ++j)
#pragma unroll
        for (int r = 0; r < 8; ++r)
          sc[i][j][r] = sc[i][j][r] * scale + mk[j];

    // ---- online softmax statistics ----
    float mnew[2][8], corr[2][8];
#pragma unroll
    for (int i = 0; i < 2; ++i)
#pragma unroll
      for (int r = 0; r < 8; ++r) {
        float v = fmaxf(sc[i][0][r], sc[i][1][r]);
        v = half16_max(v);
        const float mo = mrow[wid][i * 16 + rbase + r];
        const float mn = fmaxf(mo, v);
        mnew[i][r] = mn;
        corr[i][r] = __expf(mo - mn);
      }

    // ---- P = exp(S - m_new); LDS round-trip to reshape into A layout ----
#pragma unroll
    for (int i = 0; i < 2; ++i)
#pragma unroll
      for (int j = 0; j < 2; ++j)
#pragma unroll
        for (int r = 0; r < 8; ++r) {
          const float p = __expf(sc[i][j][r] - mnew[i][r]);
          sc[i][j][r] = p;
          Pbuf[wid][i * 16 + rbase + r][j * 16 + col] = f2bf(p);
        }

#pragma unroll
    for (int i = 0; i < 2; ++i)
#pragma unroll
      for (int r = 0; r < 8; ++r) {
        const float rsum = half16_sum(sc[i][0][r] + sc[i][1][r]);
        if (col == 0) {
          const int row = i * 16 + rbase + r;
          lrow[wid][row] = lrow[wid][row] * corr[i][r] + rsum;
          mrow[wid][row] = mnew[i][r];
        }
      }

    // ---- O = O*corr + P @ V_block (V panel in LDS) ----
#pragma unroll
    for (int i = 0; i < 2; ++i)
#pragma unroll
      for (int j = 0; j < 4; ++j)
#pragma unroll
        for (int r = 0; r < 8; ++r)
          o[i][j][r] *= corr[i][r];

    v16bf pa0 = load_frag_a(&Pbuf[wid][0][0], 40);
    v16bf pa1 = load_frag_a(&Pbuf[wid][16][0], 40);
#pragma unroll
    for (int j = 0; j < 4; ++j) {
      v16bf vb = load_frag_b(&Vbuf[buf][(j * 16) * 40], 40);
      o[0][j] = wmma_bf16(pa0, vb, o[0][j]);
      o[1][j] = wmma_bf16(pa1, vb, o[1][j]);
    }
    buf ^= 1;
  }

  // ---- finalize: O /= l ----
  float inv[2][8];
#pragma unroll
  for (int i = 0; i < 2; ++i)
#pragma unroll
    for (int r = 0; r < 8; ++r)
      inv[i][r] = 1.0f / lrow[wid][i * 16 + rbase + r];

#pragma unroll
  for (int i = 0; i < 2; ++i)
#pragma unroll
    for (int j = 0; j < 4; ++j)
#pragma unroll
      for (int r = 0; r < 8; ++r) {
        const int s = qt * 32 + i * 16 + rbase + r;
        Hout[((size_t)(b * SS + s)) * DD + h * DHH + j * 16 + col] =
            o[i][j][r] * inv[i][r];
      }
}

// ---------------------------------------------------------------------------
// residual + LayerNorm
// ---------------------------------------------------------------------------
__global__ __launch_bounds__(256)
void ln_kernel(const float* __restrict__ q, const float* __restrict__ hb,
               const float* __restrict__ gamma, const float* __restrict__ beta,
               float* __restrict__ out) {
  const int row = blockIdx.x;
  const float* xq = q  + (size_t)row * DD;
  const float* xh = hb + (size_t)row * DD;

  float s = 0.0f, s2 = 0.0f;
  for (int i = threadIdx.x; i < DD; i += 256) {
    const float x = xq[i] + xh[i];
    s += x;
    s2 += x * x;
  }
#pragma unroll
  for (int m = 1; m < 32; m <<= 1) {
    s  += __shfl_xor(s,  m, 32);
    s2 += __shfl_xor(s2, m, 32);
  }
  __shared__ float rs[8], rs2[8];
  if ((threadIdx.x & 31) == 0) {
    rs[threadIdx.x >> 5]  = s;
    rs2[threadIdx.x >> 5] = s2;
  }
  __syncthreads();
  if (threadIdx.x < 32) {
    float a  = (threadIdx.x < 8) ? rs[threadIdx.x]  : 0.0f;
    float a2 = (threadIdx.x < 8) ? rs2[threadIdx.x] : 0.0f;
#pragma unroll
    for (int m = 1; m < 8; m <<= 1) {
      a  += __shfl_xor(a,  m, 32);
      a2 += __shfl_xor(a2, m, 32);
    }
    if (threadIdx.x == 0) { rs[0] = a; rs2[0] = a2; }
  }
  __syncthreads();

  const float mean = rs[0] * (1.0f / DD);
  const float var  = rs2[0] * (1.0f / DD) - mean * mean;
  const float rinv = rsqrtf(var + 1e-12f);
  for (int i = threadIdx.x; i < DD; i += 256) {
    const float x = xq[i] + xh[i];
    out[(size_t)row * DD + i] = gamma[i] * ((x - mean) * rinv) + beta[i];
  }
}

// ---------------------------------------------------------------------------
// launch
// ---------------------------------------------------------------------------
extern "C" void kernel_launch(void* const* d_in, const int* in_sizes, int n_in,
                              void* d_out, int out_size, void* d_ws,
                              size_t ws_size, hipStream_t stream) {
  const float* query = (const float*)d_in[0];
  const float* key   = (const float*)d_in[1];
  const float* value = (const float*)d_in[2];
  const int*   mask  = (const int*)d_in[3];
  const float* Wq    = (const float*)d_in[4];
  const float* bq    = (const float*)d_in[5];
  const float* Wk    = (const float*)d_in[6];
  const float* bk    = (const float*)d_in[7];
  const float* Wv    = (const float*)d_in[8];
  const float* bv    = (const float*)d_in[9];
  const float* gamma = (const float*)d_in[10];
  const float* beta  = (const float*)d_in[11];
  (void)in_sizes; (void)n_in; (void)out_size; (void)ws_size;

  char* ws = (char*)d_ws;
  size_t off = 0;
  auto carve = [&](size_t bytes) -> char* {
    char* p = ws + off;
    off += (bytes + 255) & ~(size_t)255;
    return p;
  };

  const size_t act = (size_t)MTOT * DD;   // 4096*1024
  bf16* qb  = (bf16*)carve(act * 2);
  bf16* kb2 = (bf16*)carve(act * 2);
  bf16* vb2 = (bf16*)carve(act * 2);
  bf16* WqT = (bf16*)carve((size_t)DD * DD * 2);
  bf16* WkT = (bf16*)carve((size_t)DD * DD * 2);
  bf16* WvT = (bf16*)carve((size_t)DD * DD * 2);
  bf16* Qh  = (bf16*)carve(act * 2);      // [B,H,S,DH]
  bf16* Kh  = (bf16*)carve(act * 2);      // [B,H,S,DH]
  bf16* Vth = (bf16*)carve(act * 2);      // [B,H,DH,S]
  float* Hb = (float*)carve(act * 4);     // attention output, fp32 [B,S,D]

  const int nAct = (int)act;
  cvt_bf16_kernel<<<nAct / 256, 256, 0, stream>>>(query, qb, nAct);
  cvt_bf16_kernel<<<nAct / 256, 256, 0, stream>>>(key,   kb2, nAct);
  cvt_bf16_kernel<<<nAct / 256, 256, 0, stream>>>(value, vb2, nAct);

  const int nW = DD * DD;
  transpose_bf16_kernel<<<nW / 256, 256, 0, stream>>>(Wq, WqT);
  transpose_bf16_kernel<<<nW / 256, 256, 0, stream>>>(Wk, WkT);
  transpose_bf16_kernel<<<nW / 256, 256, 0, stream>>>(Wv, WvT);

  // (4096/128)*(1024/128) = 256 blocks of 8 waves, 128x128 tile each
  proj_kernel<<<256, 256, 0, stream>>>(qb,  WqT, bq, Qh,  0);
  proj_kernel<<<256, 256, 0, stream>>>(kb2, WkT, bk, Kh,  0);
  proj_kernel<<<256, 256, 0, stream>>>(vb2, WvT, bv, Vth, 1);

  // 32 (b,h) x 16 blocks x 4 waves (q-tiles of 32)
  attn_kernel<<<512, 128, 0, stream>>>(Qh, Kh, Vth, mask, Hb);

  ln_kernel<<<MTOT, 256, 0, stream>>>(query, Hb, gamma, beta, (float*)d_out);
}
